// Decoder_51582557225711
// MI455X (gfx1250) — matrix operand order repaired and checked by reading, not compiled, not run on hardware
//
#include <hip/hip_runtime.h>
#include <hip/hip_bf16.h>

// ---------------------------------------------------------------------------
// TinyLlama-class prefill forward for gfx1250 (MI455X).
// - GEMMs: bf16 x bf16 -> f32 via v_wmma_f32_16x16x32_bf16 (16x16x32 native).
// - LDS staging: double-buffered; straight-copy tiles use CDNA5 async
//   global->LDS DMA (global_load_async_to_lds_b128 + s_wait_asynccnt);
//   fp32 weight tiles are converted to bf16 on the fly (sync path) but still
//   overlap with WMMA work via the double buffer.
// - Output = pre-RoPE K/V caches [2, L, 1, H, S, HD] fp32, written by the
//   fused RoPE kernel directly into d_out.
// ---------------------------------------------------------------------------

#define S_LEN 2048
#define EMB   2048
#define NH    16
#define HD    128
#define FFN   5632
#define NL    4

typedef __attribute__((ext_vector_type(16))) __bf16 v16bf;
typedef __attribute__((ext_vector_type(8)))  __bf16 v8bf;
typedef __attribute__((ext_vector_type(8)))  float  v8f;

__device__ __forceinline__ __bf16 f2bf(float f) {
    union { float f; unsigned u; } a; a.f = f;
    unsigned r = a.u + 0x7FFFu + ((a.u >> 16) & 1u);   // round-to-nearest-even
    unsigned short h = (unsigned short)(r >> 16);
    __bf16 out; __builtin_memcpy(&out, &h, 2);
    return out;
}

// Async 16-byte global->LDS copy (CDNA5 TDM-adjacent path, ASYNCcnt-tracked).
__device__ __forceinline__ void async_copy_b128(const void* gptr, void* lptr) {
    unsigned lds_off = (unsigned)(size_t)lptr;          // addrspace(3) offset
    asm volatile("global_load_async_to_lds_b128 %0, %1, off"
                 :: "v"(lds_off), "v"(gptr) : "memory");
}
__device__ __forceinline__ void wait_async() {
    asm volatile("s_wait_asynccnt 0x0" ::: "memory");
}

// ---------------------------------------------------------------------------
// Tiled WMMA GEMM:  C[M,N] = A[M,K] (bf16) @ op(B)  (TB = float | __bf16)
//   TRANSB=false : B is [K,N] row-major
//   TRANSB=true  : B is [N,K] row-major (C = A @ B^T)
// Block tile 128x128, BK=32, 256 threads = 8 waves (2x4), wave tile 64x32.
// Double-buffered LDS; A (and B when TRANSB) staged by async DMA.
// ---------------------------------------------------------------------------
template <typename TB, bool TRANSB, bool OUT_BF16>
__global__ __launch_bounds__(256)
void gemm_wmma_kernel(const __bf16* __restrict__ A, const TB* __restrict__ B,
                      void* __restrict__ Cout,
                      int M, int N, int K, int lda, int ldb, int ldc)
{
    __shared__ alignas(16) __bf16 Alds[2][128][32];   // [buf][m][k]
    __shared__ alignas(16) __bf16 Blds[2][128][32];   // [buf][n][k] (k-contig)

    const int tid  = threadIdx.x;
    const int wid  = tid >> 5;
    const int lane = tid & 31;
    const int wm   = wid >> 2;          // 0..1 -> 64-row slab
    const int wn   = wid & 3;           // 0..3 -> 32-col slab
    const int hi   = lane >> 4;         // wave half
    const int l15  = lane & 15;
    const int m0   = blockIdx.y * 128;
    const int n0   = blockIdx.x * 128;

    auto stageA = [&](int buf, int k0) {
#pragma unroll
        for (int it = 0; it < 2; ++it) {
            int chunk = tid + it * 256;              // 0..511
            int r  = chunk >> 2;                     // 0..127
            int kc = (chunk & 3) * 8;                // 0,8,16,24
            async_copy_b128(&A[(size_t)(m0 + r) * lda + k0 + kc],
                            &Alds[buf][r][kc]);
        }
    };
    auto stageB = [&](int buf, int k0) {
        if constexpr (TRANSB) {
            // B[N,K] bf16 row-major: straight async copy
#pragma unroll
            for (int it = 0; it < 2; ++it) {
                int chunk = tid + it * 256;
                int r  = chunk >> 2;
                int kc = (chunk & 3) * 8;
                async_copy_b128(&B[(size_t)(n0 + r) * ldb + k0 + kc],
                                &Blds[buf][r][kc]);
            }
        } else if constexpr (sizeof(TB) == 4) {
            // fp32 weights [K,N]: convert to bf16 while transposing into LDS
#pragma unroll
            for (int it = 0; it < 4; ++it) {
                int chunk = tid + it * 256;          // 0..1023
                int kk = chunk >> 5;                 // 0..31
                int nc = (chunk & 31) * 4;           // 0..124
                const float4 f =
                    *(const float4*)&B[(size_t)(k0 + kk) * ldb + n0 + nc];
                Blds[buf][nc + 0][kk] = f2bf(f.x);
                Blds[buf][nc + 1][kk] = f2bf(f.y);
                Blds[buf][nc + 2][kk] = f2bf(f.z);
                Blds[buf][nc + 3][kk] = f2bf(f.w);
            }
        } else {
            // bf16 B [K,N] row-major (probs @ V): scatter transpose
#pragma unroll
            for (int it = 0; it < 2; ++it) {
                int chunk = tid + it * 256;          // 0..511
                int kk = chunk >> 4;                 // 0..31
                int nc = (chunk & 15) * 8;           // 0..120
                v8bf b = *(const v8bf*)&B[(size_t)(k0 + kk) * ldb + n0 + nc];
#pragma unroll
                for (int j = 0; j < 8; ++j) Blds[buf][nc + j][kk] = b[j];
            }
        }
    };

    v8f acc[4][2];
#pragma unroll
    for (int mt = 0; mt < 4; ++mt)
#pragma unroll
        for (int nt = 0; nt < 2; ++nt) { v8f z = {}; acc[mt][nt] = z; }

    // prologue: stage first K-tile into buffer 0
    stageA(0, 0);
    stageB(0, 0);

    int cur = 0;
    for (int k0 = 0; k0 < K; k0 += 32) {
        wait_async();        // this wave's DMA into `cur` complete
        __syncthreads();     // + s_wait_dscnt: all staging of `cur` visible

        if (k0 + 32 < K) {   // prefetch next tile into the other buffer
            stageA(cur ^ 1, k0 + 32);
            stageB(cur ^ 1, k0 + 32);
        }

        // ---- build fragments per documented 16-bit WMMA layouts ----
        v16bf afr[4];
#pragma unroll
        for (int mt = 0; mt < 4; ++mt) {
            int row = wm * 64 + mt * 16 + l15;
            union { v16bf v; v8bf h[2]; } u;
            u.h[0] = *(const v8bf*)&Alds[cur][row][hi * 8];      // K 0-7 / 8-15
            u.h[1] = *(const v8bf*)&Alds[cur][row][16 + hi * 8]; // K 16-23 / 24-31
            afr[mt] = u.v;
        }
        v16bf bfr[2];
#pragma unroll
        for (int nt = 0; nt < 2; ++nt) {
            int col = wn * 32 + nt * 16 + l15;
            union { v16bf v; v8bf h[2]; } u;
            u.h[0] = *(const v8bf*)&Blds[cur][col][hi * 8];      // K 0-7 / 8-15
            u.h[1] = *(const v8bf*)&Blds[cur][col][16 + hi * 8]; // K 16-23 / 24-31
            bfr[nt] = u.v;
        }
#pragma unroll
        for (int mt = 0; mt < 4; ++mt)
#pragma unroll
            for (int nt = 0; nt < 2; ++nt)
                acc[mt][nt] = __builtin_amdgcn_wmma_f32_16x16x32_bf16(
                    false, afr[mt], false, bfr[nt],
                    (short)0, acc[mt][nt], false, false);

        cur ^= 1;
    }

    // ---- store C (16x16 f32 frag: col = lane&15, rows r + 8*(lane>>4)) ----
#pragma unroll
    for (int mt = 0; mt < 4; ++mt)
#pragma unroll
        for (int nt = 0; nt < 2; ++nt) {
            int col = n0 + wn * 32 + nt * 16 + l15;
            int rb  = m0 + wm * 64 + mt * 16 + hi * 8;
            if constexpr (OUT_BF16) {
                __bf16* C = (__bf16*)Cout;
#pragma unroll
                for (int r = 0; r < 8; ++r)
                    C[(size_t)(rb + r) * ldc + col] = f2bf(acc[mt][nt][r]);
            } else {
                float* C = (float*)Cout;
#pragma unroll
                for (int r = 0; r < 8; ++r)
                    C[(size_t)(rb + r) * ldc + col] = acc[mt][nt][r];
            }
        }
}

// ---------------------------------------------------------------------------
// Elementwise / reduction kernels
// ---------------------------------------------------------------------------
__global__ __launch_bounds__(256)
void embed_kernel(const int* __restrict__ ids, const float* __restrict__ emb,
                  float* __restrict__ x)
{
    int s = blockIdx.x;
    size_t id = (size_t)ids[s];
    for (int c = threadIdx.x; c < EMB; c += 256)
        x[(size_t)s * EMB + c] = emb[id * EMB + c];
}

__global__ __launch_bounds__(256)
void rmsnorm_kernel(const float* __restrict__ x, const float* __restrict__ w,
                    __bf16* __restrict__ out)
{
    __shared__ float red[256];
    int row = blockIdx.x, t = threadIdx.x;
    const float* xr = x + (size_t)row * EMB;
    float ss = 0.f;
    for (int c = t; c < EMB; c += 256) { float v = xr[c]; ss += v * v; }
    red[t] = ss; __syncthreads();
    for (int st = 128; st > 0; st >>= 1) {
        if (t < st) red[t] += red[t + st];
        __syncthreads();
    }
    float scale = rsqrtf(red[0] / (float)EMB + 1e-5f);
    for (int c = t; c < EMB; c += 256)
        out[(size_t)row * EMB + c] = f2bf(xr[c] * scale * w[c]);
}

// RoPE on q,k -> bf16 [H][S][HD]; v -> bf16 [H][S][HD];
// pre-RoPE k,v -> fp32 straight into the output KV cache.
__global__ __launch_bounds__(128)
void rope_qkv_kernel(const float* __restrict__ q, const float* __restrict__ k,
                     const float* __restrict__ v,
                     __bf16* __restrict__ qr, __bf16* __restrict__ kr,
                     __bf16* __restrict__ vb,
                     float* __restrict__ kout, float* __restrict__ vout)
{
    int s = blockIdx.x, h = blockIdx.y, d = threadIdx.x;
    size_t iin  = (size_t)s * (NH * HD) + (size_t)h * HD + d;
    size_t iout = ((size_t)h * S_LEN + s) * HD + d;

    float qv = q[iin], kv = k[iin], vv = v[iin];
    int p = d & 63;                                  // pair index
    float invf = __expf(-(float)p * (9.210340371976184f / 64.0f)); // theta^-2p/HD
    float ang = (float)s * invf;
    float c = __cosf(ang), si = __sinf(ang);
    float qh = (d < 64) ? -q[iin + 64] : q[iin - 64];
    float kh = (d < 64) ? -k[iin + 64] : k[iin - 64];

    qr[iout] = f2bf(qv * c + qh * si);
    kr[iout] = f2bf(kv * c + kh * si);
    vb[iout] = f2bf(vv);
    kout[iout] = kv;     // pre-RoPE cache
    vout[iout] = vv;
}

__global__ __launch_bounds__(256)
void softmax_causal_kernel(const float* __restrict__ sc, __bf16* __restrict__ p,
                           float scale)
{
    __shared__ float red[256];
    int row = blockIdx.x, t = threadIdx.x;
    const float* sr = sc + (size_t)row * S_LEN;
    __bf16* pr = p + (size_t)row * S_LEN;
    int valid = row + 1;

    float mx = -1e30f;
    for (int c = t; c < valid; c += 256) mx = fmaxf(mx, sr[c] * scale);
    red[t] = mx; __syncthreads();
    for (int st = 128; st > 0; st >>= 1) {
        if (t < st) red[t] = fmaxf(red[t], red[t + st]);
        __syncthreads();
    }
    mx = red[0]; __syncthreads();

    float sum = 0.f;
    for (int c = t; c < valid; c += 256) sum += __expf(sr[c] * scale - mx);
    red[t] = sum; __syncthreads();
    for (int st = 128; st > 0; st >>= 1) {
        if (t < st) red[t] += red[t + st];
        __syncthreads();
    }
    float inv = 1.0f / red[0];

    for (int c = t; c < S_LEN; c += 256) {
        float pv = (c < valid) ? __expf(sr[c] * scale - mx) * inv : 0.0f;
        pr[c] = f2bf(pv);
    }
}

__global__ __launch_bounds__(256)
void silu_mul_kernel(const float* __restrict__ g, const float* __restrict__ u,
                     __bf16* __restrict__ m, int n)
{
    int i = blockIdx.x * 256 + threadIdx.x;
    if (i < n) {
        float gv = g[i];
        m[i] = f2bf(gv / (1.0f + __expf(-gv)) * u[i]);
    }
}

__global__ __launch_bounds__(256)
void residual_add_kernel(float* __restrict__ x, const float* __restrict__ t, int n)
{
    int i = blockIdx.x * 256 + threadIdx.x;
    if (i < n) x[i] += t[i];
}

// ---------------------------------------------------------------------------
// Host orchestration
// ---------------------------------------------------------------------------
extern "C" void kernel_launch(void* const* d_in, const int* in_sizes, int n_in,
                              void* d_out, int out_size, void* d_ws, size_t ws_size,
                              hipStream_t stream)
{
    (void)in_sizes; (void)n_in; (void)out_size; (void)ws_size;

    const int*   ids   = (const int*)  d_in[0];
    const float* emb   = (const float*)d_in[1];
    const float* Wq    = (const float*)d_in[2];
    const float* Wk    = (const float*)d_in[3];
    const float* Wv    = (const float*)d_in[4];
    const float* Wo    = (const float*)d_in[5];
    const float* Wg    = (const float*)d_in[6];
    const float* Wu    = (const float*)d_in[7];
    const float* Wd    = (const float*)d_in[8];
    const float* ln1   = (const float*)d_in[9];
    const float* ln2   = (const float*)d_in[10];
    float*       out   = (float*)d_out;

    // ---- workspace carve (256B aligned) ----
    size_t off = 0;
    auto carve = [&](size_t bytes) -> void* {
        void* p = (char*)d_ws + off;
        off += (bytes + 255) & ~(size_t)255;
        return p;
    };
    float*  x    = (float*) carve((size_t)S_LEN * EMB * 4);
    __bf16* hbf  = (__bf16*)carve((size_t)S_LEN * EMB * 2);
    float*  qf   = (float*) carve((size_t)S_LEN * (NH * HD) * 4);
    float*  kf   = (float*) carve((size_t)S_LEN * (NH * HD) * 4);
    float*  vf   = (float*) carve((size_t)S_LEN * (NH * HD) * 4);
    __bf16* qr   = (__bf16*)carve((size_t)NH * S_LEN * HD * 2);
    __bf16* kr   = (__bf16*)carve((size_t)NH * S_LEN * HD * 2);
    __bf16* vb   = (__bf16*)carve((size_t)NH * S_LEN * HD * 2);
    __bf16* att  = (__bf16*)carve((size_t)S_LEN * (NH * HD) * 2);
    float*  sc   = (float*) carve((size_t)S_LEN * S_LEN * 4);
    __bf16* pb   = (__bf16*)carve((size_t)S_LEN * S_LEN * 2);
    float*  gf   = (float*) carve((size_t)S_LEN * FFN * 4);
    float*  uf   = (float*) carve((size_t)S_LEN * FFN * 4);
    __bf16* mbf  = (__bf16*)carve((size_t)S_LEN * FFN * 2);

    const float att_scale = 0.08838834764831845f;   // 1/sqrt(128)
    const size_t kv_blk = (size_t)NH * S_LEN * HD;  // per-(tensor,layer) block

    dim3 blk(256);

    // embedding gather
    embed_kernel<<<S_LEN, blk, 0, stream>>>(ids, emb, x);

    for (int l = 0; l < NL; ++l) {
        const float* wq = Wq + (size_t)l * EMB * (NH * HD);
        const float* wk = Wk + (size_t)l * EMB * (NH * HD);
        const float* wv = Wv + (size_t)l * EMB * (NH * HD);
        const float* wo = Wo + (size_t)l * (NH * HD) * EMB;
        const float* wg = Wg + (size_t)l * EMB * FFN;
        const float* wu = Wu + (size_t)l * EMB * FFN;
        const float* wd = Wd + (size_t)l * FFN * EMB;

        // --- attention block ---
        rmsnorm_kernel<<<S_LEN, blk, 0, stream>>>(x, ln1 + (size_t)l * EMB, hbf);

        dim3 gQKV(EMB / 128, S_LEN / 128);
        gemm_wmma_kernel<float, false, false><<<gQKV, blk, 0, stream>>>(
            hbf, wq, qf, S_LEN, NH * HD, EMB, EMB, NH * HD, NH * HD);
        gemm_wmma_kernel<float, false, false><<<gQKV, blk, 0, stream>>>(
            hbf, wk, kf, S_LEN, NH * HD, EMB, EMB, NH * HD, NH * HD);
        gemm_wmma_kernel<float, false, false><<<gQKV, blk, 0, stream>>>(
            hbf, wv, vf, S_LEN, NH * HD, EMB, EMB, NH * HD, NH * HD);

        rope_qkv_kernel<<<dim3(S_LEN, NH), dim3(128), 0, stream>>>(
            qf, kf, vf, qr, kr, vb,
            out + (size_t)l * kv_blk,            // keys  [0,l]
            out + (size_t)(NL + l) * kv_blk);    // vals  [1,l]

        for (int h = 0; h < NH; ++h) {
            const __bf16* qh = qr + (size_t)h * S_LEN * HD;
            const __bf16* kh = kr + (size_t)h * S_LEN * HD;
            const __bf16* vh = vb + (size_t)h * S_LEN * HD;

            // scores = Qr @ Kr^T   (M=S, N=S, K=HD)
            dim3 gQK(S_LEN / 128, S_LEN / 128);
            gemm_wmma_kernel<__bf16, true, false><<<gQK, blk, 0, stream>>>(
                qh, kh, sc, S_LEN, S_LEN, HD, HD, HD, S_LEN);

            softmax_causal_kernel<<<S_LEN, blk, 0, stream>>>(sc, pb, att_scale);

            // O_h = P @ V_h  (M=S, N=HD, K=S), bf16 out at column offset h*HD
            dim3 gPV(HD / 128, S_LEN / 128);
            gemm_wmma_kernel<__bf16, false, true><<<gPV, blk, 0, stream>>>(
                pb, vh, att + (size_t)h * HD, S_LEN, HD, S_LEN,
                S_LEN, HD, NH * HD);
        }

        // o = att @ Wo ; x += o
        gemm_wmma_kernel<float, false, false><<<gQKV, blk, 0, stream>>>(
            att, wo, qf, S_LEN, EMB, NH * HD, NH * HD, EMB, EMB);
        residual_add_kernel<<<(S_LEN * EMB) / 256, blk, 0, stream>>>(
            x, qf, S_LEN * EMB);

        // --- MLP block ---
        rmsnorm_kernel<<<S_LEN, blk, 0, stream>>>(x, ln2 + (size_t)l * EMB, hbf);

        dim3 gFF(FFN / 128, S_LEN / 128);
        gemm_wmma_kernel<float, false, false><<<gFF, blk, 0, stream>>>(
            hbf, wg, gf, S_LEN, FFN, EMB, EMB, FFN, FFN);
        gemm_wmma_kernel<float, false, false><<<gFF, blk, 0, stream>>>(
            hbf, wu, uf, S_LEN, FFN, EMB, EMB, FFN, FFN);

        silu_mul_kernel<<<(S_LEN * FFN) / 256, blk, 0, stream>>>(
            gf, uf, mbf, S_LEN * FFN);

        gemm_wmma_kernel<float, false, false><<<gQKV, blk, 0, stream>>>(
            mbf, wd, qf, S_LEN, EMB, FFN, FFN, EMB, EMB);
        residual_add_kernel<<<(S_LEN * EMB) / 256, blk, 0, stream>>>(
            x, qf, S_LEN * EMB);
    }
    // final rms_norm in reference is computed but not returned -> skip.
}
